// MambaActor_80058190398116
// MI455X (gfx1250) — matrix operand order repaired and checked by reading, not compiled, not run on hardware
//
#include <hip/hip_runtime.h>
#include <hip/hip_bf16.h>
#include <math.h>

// ---------------------------------------------------------------------------
// Types for CDNA5 WMMA (wave32)
// ---------------------------------------------------------------------------
typedef unsigned short ushort_t;
typedef __attribute__((ext_vector_type(16))) __bf16        v16bf;
typedef __attribute__((ext_vector_type(8)))  float         v8f;
typedef __attribute__((ext_vector_type(8)))  unsigned short us8;

union FragAB {
    us8   u[2];   // two 16-byte halves
    v16bf bf;     // 16 bf16 operand fragment
};

__device__ __forceinline__ ushort_t f2bf(float f) {
    unsigned u = __float_as_uint(f);
    u += 0x7FFFu + ((u >> 16) & 1u);        // round-to-nearest-even
    return (ushort_t)(u >> 16);
}

// ---------------------------------------------------------------------------
// fp32 -> bf16 conversion (weights / input state)
// ---------------------------------------------------------------------------
__global__ void cvt_bf16_kernel(const float* __restrict__ in,
                                ushort_t* __restrict__ out, size_t n) {
    size_t i      = (size_t)blockIdx.x * blockDim.x + threadIdx.x;
    size_t stride = (size_t)gridDim.x * blockDim.x;
    for (; i < n; i += stride) out[i] = f2bf(in[i]);
}

// ---------------------------------------------------------------------------
// Generic bf16 WMMA GEMM:  C[M,N] = act(A[M,K] * W[N,K]^T + bias)
//   A: bf16 row-major, leading dim lda
//   W: bf16 [N,K] row-major (i.e. B^T), leading dim ldw
//   Outputs: optional fp32 (Cf) and/or bf16 (Cb), both leading dim N.
//   ACT: 0=none 1=relu 2=softplus
// Block: 256 threads = 8 waves (4 rows x 2 cols), block tile 128 x 128,
// wave tile 32 x 64 (2 A-frags x 4 B-frags = 8 WMMA per K-step).
// Loads are branch-free: out-of-range columns clamp to N-1 (stores guarded).
// M must be a multiple of 128, K a multiple of 32.
// ---------------------------------------------------------------------------
template<int ACT>
__global__ __launch_bounds__(256)
void gemm_bf16_wmma(const ushort_t* __restrict__ A, int lda,
                    const ushort_t* __restrict__ W, int ldw,
                    const float* __restrict__ bias,
                    float* __restrict__ Cf, ushort_t* __restrict__ Cb,
                    int M, int N, int K)
{
    const int lane = threadIdx.x & 31;
    const int wid  = threadIdx.x >> 5;
    const int wm   = wid & 3;          // 0..3 -> 32-row slice
    const int wn   = wid >> 2;         // 0..1 -> 64-col slice
    const int half = lane >> 4;        // 0/1
    const int l16  = lane & 15;

    const int baseM = blockIdx.y * 128 + wm * 32;
    const int baseN = blockIdx.x * 128 + wn * 64;

    v8f c[2][4] = {};

    const char* Ab = (const char*)A;
    const char* Wb = (const char*)W;

    // 32-bit per-lane byte offsets against uniform bases (SADDR-friendly).
    unsigned aoff[2];
    #pragma unroll
    for (int i = 0; i < 2; ++i)
        aoff[i] = (unsigned)(((baseM + i * 16 + l16) * lda + 8 * half) * 2);

    int      ncol[4];
    unsigned boff[4];
    #pragma unroll
    for (int j = 0; j < 4; ++j) {
        const int n  = baseN + j * 16 + l16;
        ncol[j]      = n;
        const int nc = (n < N) ? n : (N - 1);          // clamp: loads stay valid
        boff[j]      = (unsigned)((nc * ldw + 8 * half) * 2);
    }

    const int Kb = K * 2;   // K in bytes
    for (int kk = 0; kk < Kb; kk += 64) {               // 32 K-elems per step
        FragAB a[2], b[4];
        // A fragment (16-bit A 16x32): lanes 0-15 hold K {0..7,16..23},
        // lanes 16-31 hold K {8..15,24..31} of row (lane&15).
        #pragma unroll
        for (int i = 0; i < 2; ++i) {
            a[i].u[0] = *(const us8*)(Ab + (size_t)(aoff[i] + kk));
            a[i].u[1] = *(const us8*)(Ab + (size_t)(aoff[i] + kk + 32));
        }
        // B fragment (16-bit B 32x16): lane holds one column, K {0..7,16..23}
        // for lanes 0-15 and K {8..15,24..31} for lanes 16-31 (matches A).
        #pragma unroll
        for (int j = 0; j < 4; ++j) {
            b[j].u[0] = *(const us8*)(Wb + (size_t)(boff[j] + kk));
            b[j].u[1] = *(const us8*)(Wb + (size_t)(boff[j] + kk + 32));
        }

        if (kk + 128 < Kb) {    // global_prefetch_b8 hints two steps ahead
            __builtin_prefetch(Ab + (size_t)(aoff[0] + kk + 128), 0, 1);
            __builtin_prefetch(Ab + (size_t)(aoff[1] + kk + 128), 0, 1);
            __builtin_prefetch(Wb + (size_t)(boff[0] + kk + 128), 0, 1);
            __builtin_prefetch(Wb + (size_t)(boff[2] + kk + 128), 0, 1);
        }

        #pragma unroll
        for (int i = 0; i < 2; ++i)
            #pragma unroll
            for (int j = 0; j < 4; ++j)
                c[i][j] = __builtin_amdgcn_wmma_f32_16x16x32_bf16(
                    false, a[i].bf, false, b[j].bf, (short)0, c[i][j],
                    false, false);
    }

    // Epilogue. C/D layout: lane(0-15) holds N=lane, rows r+8*half per VGPR r.
    #pragma unroll
    for (int j = 0; j < 4; ++j) {
        const int n = ncol[j];
        if (n >= N) continue;
        const float bv = bias ? bias[n] : 0.0f;
        #pragma unroll
        for (int i = 0; i < 2; ++i) {
            const int m0 = baseM + i * 16 + 8 * half;
            #pragma unroll
            for (int r = 0; r < 8; ++r) {
                float v = c[i][j][r] + bv;
                if (ACT == 1)      v = fmaxf(v, 0.0f);
                else if (ACT == 2) v = (v > 20.0f) ? v : log1pf(__expf(v));
                const size_t o = (size_t)(m0 + r) * N + n;
                if (Cf) Cf[o] = v;
                if (Cb) Cb[o] = f2bf(v);
            }
        }
    }
}

// ---------------------------------------------------------------------------
// Causal depthwise conv (DC=4) + bias + silu over xs = xz[:, 0:2048].
// Writes fp32 (for scan) and bf16 (for x_proj GEMM).
// ---------------------------------------------------------------------------
__global__ __launch_bounds__(256)
void conv_silu_kernel(const float* __restrict__ xz,
                      const float* __restrict__ cw,
                      const float* __restrict__ cb,
                      float* __restrict__ xs_f,
                      ushort_t* __restrict__ xs_b)
{
    const size_t idx = (size_t)blockIdx.x * 256 + threadIdx.x;
    if (idx >= (size_t)16384 * 2048) return;
    const int    d   = (int)(idx & 2047);
    const size_t row = idx >> 11;            // b*L + l
    const int    l   = (int)(row & 2047);    // L = 2048
    float acc = cb[d];
    #pragma unroll
    for (int j = 0; j < 4; ++j) {
        const int ll = l - 3 + j;
        if (ll >= 0) acc += cw[d * 4 + j] * xz[(row - 3 + j) * 4096 + d];
    }
    const float s = acc / (1.0f + __expf(-acc));   // silu
    xs_f[idx] = s;
    xs_b[idx] = f2bf(s);
}

// ---------------------------------------------------------------------------
// Selective scan. Lane layout: 16 lanes = 16 states of one channel (b,d);
// wave32 holds 2 channels. Block 256 thr = 16 channels (consecutive d).
// Fuses D-skip, silu(z) gate and bf16 conversion.
// ---------------------------------------------------------------------------
__global__ __launch_bounds__(256)
void scan_kernel(const float* __restrict__ delta,
                 const float* __restrict__ xs,
                 const float* __restrict__ xdbl,   // [M,96]: dt|B|C
                 const float* __restrict__ xz,     // z = cols 2048..4095
                 const float* __restrict__ A_log,
                 const float* __restrict__ Dp,
                 ushort_t* __restrict__ y_b)
{
    const int s   = threadIdx.x & 15;
    const int grp = threadIdx.x >> 4;
    const int d   = blockIdx.x * 16 + grp;
    const int b   = blockIdx.y;
    const float Av = -__expf(A_log[d * 16 + s]);
    const float Dv = Dp[d];
    float h = 0.0f;
    const size_t rowbase = (size_t)b * 2048;
    for (int l = 0; l < 2048; ++l) {
        const size_t row = rowbase + l;
        const float dt = delta[row * 2048 + d];     // broadcast across 16 lanes
        const float xv = xs[row * 2048 + d];
        const float Bv = xdbl[row * 96 + 64 + s];
        const float Cv = xdbl[row * 96 + 80 + s];
        h = h * __expf(dt * Av) + (dt * Bv) * xv;
        float y = h * Cv;
        y += __shfl_xor(y, 1, 16);
        y += __shfl_xor(y, 2, 16);
        y += __shfl_xor(y, 4, 16);
        y += __shfl_xor(y, 8, 16);
        if (s == 0) {
            const float z  = xz[row * 4096 + 2048 + d];
            const float yy = (y + xv * Dv) * (z / (1.0f + __expf(-z)));
            y_b[row * 2048 + d] = f2bf(yy);
        }
    }
}

// ---------------------------------------------------------------------------
// Row-wise argmax of 256 logits (argmax(softmax(x)) == argmax(x)).
// One wave per row; first-occurrence tie-break (lowest index).
// ---------------------------------------------------------------------------
__global__ __launch_bounds__(256)
void argmax_kernel(const float* __restrict__ logits, int* __restrict__ out)
{
    const int lane = threadIdx.x & 31;
    const int wid  = threadIdx.x >> 5;
    const size_t row = (size_t)blockIdx.x * 8 + wid;
    const float* p = logits + row * 256;
    float bv = -INFINITY;
    int   bi = 0x7FFFFFFF;
    #pragma unroll
    for (int i = 0; i < 8; ++i) {
        const int   idx = lane + i * 32;
        const float v   = p[idx];
        if (v > bv || (v == bv && idx < bi)) { bv = v; bi = idx; }
    }
    #pragma unroll
    for (int m = 16; m >= 1; m >>= 1) {
        const float ov = __shfl_xor(bv, m, 32);
        const int   oi = __shfl_xor(bi, m, 32);
        if (ov > bv || (ov == bv && oi < bi)) { bv = ov; bi = oi; }
    }
    if (lane == 0) out[row] = bi;
}

// ---------------------------------------------------------------------------
// Launcher
// ---------------------------------------------------------------------------
extern "C" void kernel_launch(void* const* d_in, const int* in_sizes, int n_in,
                              void* d_out, int out_size, void* d_ws, size_t ws_size,
                              hipStream_t stream)
{
    const float* state      = (const float*)d_in[0];
    const float* fc1_w      = (const float*)d_in[1];
    const float* fc1_b      = (const float*)d_in[2];
    const float* in_proj_w  = (const float*)d_in[3];
    const float* conv_w     = (const float*)d_in[4];
    const float* conv_b     = (const float*)d_in[5];
    const float* x_proj_w   = (const float*)d_in[6];
    const float* dt_proj_w  = (const float*)d_in[7];
    const float* dt_proj_b  = (const float*)d_in[8];
    const float* A_log      = (const float*)d_in[9];
    const float* D_param    = (const float*)d_in[10];
    const float* out_proj_w = (const float*)d_in[11];
    const float* fc3_w      = (const float*)d_in[12];
    const float* fc3_b      = (const float*)d_in[13];

    constexpr size_t M = 16384;   // B*L
    char* ws = (char*)d_ws;
    size_t off = 0;
    auto alloc = [&](size_t bytes) -> size_t {
        off = (off + 255) & ~(size_t)255;
        size_t o = off; off += bytes; return o;
    };

    ushort_t* state_bf   = (ushort_t*)(ws + alloc(M * 512 * 2));
    ushort_t* fc1_bf     = (ushort_t*)(ws + alloc((size_t)1024 * 512 * 2));
    ushort_t* inproj_bf  = (ushort_t*)(ws + alloc((size_t)4096 * 1024 * 2));
    ushort_t* xproj_bf   = (ushort_t*)(ws + alloc((size_t)96 * 2048 * 2));
    ushort_t* dtproj_bf  = (ushort_t*)(ws + alloc((size_t)2048 * 64 * 2));
    ushort_t* outproj_bf = (ushort_t*)(ws + alloc((size_t)1024 * 2048 * 2));
    ushort_t* fc3_bf     = (ushort_t*)(ws + alloc((size_t)256 * 1024 * 2));
    ushort_t* x_bf       = (ushort_t*)(ws + alloc(M * 1024 * 2));
    size_t    xz_off     = alloc(M * 4096 * 4);
    float*    xz         = (float*)(ws + xz_off);
    float*    xs_f       = (float*)(ws + alloc(M * 2048 * 4));
    ushort_t* xs_bf      = (ushort_t*)(ws + alloc(M * 2048 * 2));
    float*    x_dbl      = (float*)(ws + alloc(M * 96 * 4));
    ushort_t* x_dbl_bf   = (ushort_t*)(ws + alloc(M * 96 * 2));
    float*    delta      = (float*)(ws + alloc(M * 2048 * 4));
    ushort_t* y_bf       = (ushort_t*)(ws + alloc(M * 2048 * 2));
    ushort_t* m_bf       = (ushort_t*)(ws + alloc(M * 1024 * 2));
    // logits alias the xz region (z fully consumed by the scan before fc3)
    float*    logits     = (float*)(ws + xz_off);

    auto cvt = [&](const float* in, ushort_t* out, size_t n) {
        unsigned blocks = (unsigned)((n + 255) / 256);
        cvt_bf16_kernel<<<blocks, 256, 0, stream>>>(in, out, n);
    };
    cvt(state,      state_bf,   M * 512);
    cvt(fc1_w,      fc1_bf,     (size_t)1024 * 512);
    cvt(in_proj_w,  inproj_bf,  (size_t)4096 * 1024);
    cvt(x_proj_w,   xproj_bf,   (size_t)96 * 2048);
    cvt(dt_proj_w,  dtproj_bf,  (size_t)2048 * 64);
    cvt(out_proj_w, outproj_bf, (size_t)1024 * 2048);
    cvt(fc3_w,      fc3_bf,     (size_t)256 * 1024);

    const dim3 blk(256);
    auto grid = [&](int N) { return dim3((unsigned)((N + 127) / 128), (unsigned)(M / 128)); };

    // fc1: relu(state @ fc1^T + b) -> x_bf
    gemm_bf16_wmma<1><<<grid(1024), blk, 0, stream>>>(
        state_bf, 512, fc1_bf, 512, fc1_b, nullptr, x_bf, (int)M, 1024, 512);

    // in_proj: xz = x @ in_proj^T (fp32; xs|z)
    gemm_bf16_wmma<0><<<grid(4096), blk, 0, stream>>>(
        x_bf, 1024, inproj_bf, 1024, nullptr, xz, nullptr, (int)M, 4096, 1024);

    // depthwise causal conv + silu
    conv_silu_kernel<<<(unsigned)((M * 2048) / 256), blk, 0, stream>>>(
        xz, conv_w, conv_b, xs_f, xs_bf);

    // x_proj: x_dbl = xs @ x_proj^T (fp32 for B/C, bf16 for dt GEMM)
    gemm_bf16_wmma<0><<<grid(96), blk, 0, stream>>>(
        xs_bf, 2048, xproj_bf, 2048, nullptr, x_dbl, x_dbl_bf, (int)M, 96, 2048);

    // dt_proj: delta = softplus(dt @ dt_proj^T + b)
    gemm_bf16_wmma<2><<<grid(2048), blk, 0, stream>>>(
        x_dbl_bf, 96, dtproj_bf, 64, dt_proj_b, delta, nullptr, (int)M, 2048, 64);

    // selective scan + D skip + silu(z) gate -> y_bf
    scan_kernel<<<dim3(2048 / 16, 8), blk, 0, stream>>>(
        delta, xs_f, x_dbl, xz, A_log, D_param, y_bf);

    // out_proj + relu -> m_bf
    gemm_bf16_wmma<1><<<grid(1024), blk, 0, stream>>>(
        y_bf, 2048, outproj_bf, 2048, nullptr, nullptr, m_bf, (int)M, 1024, 2048);

    // fc3 + relu -> logits
    gemm_bf16_wmma<1><<<grid(256), blk, 0, stream>>>(
        m_bf, 1024, fc3_bf, 1024, fc3_b, logits, nullptr, (int)M, 256, 1024);

    // argmax -> int32 actions
    argmax_kernel<<<(unsigned)(M / 8), blk, 0, stream>>>(logits, (int*)d_out);
}
